// ICOHPPredictor_47004122087426
// MI455X (gfx1250) — matrix operand order repaired and checked by reading, not compiled, not run on hardware
//
#include <hip/hip_runtime.h>
#include <math.h>

// ---------------- constants ----------------
#define NNODES 20000
#define NEDGES 320000
#define DA 92
#define DAP 96          // padded K for atom features
#define DE 128
#define DN 256
#define DIN_E 224       // 2*92 + 40
#define DH 640          // head input: 128 + 256 + 256
#define NLAYERS 4

typedef __attribute__((ext_vector_type(16))) __bf16 v16bf;
typedef __attribute__((ext_vector_type(8)))  __bf16 v8bf;
typedef __attribute__((ext_vector_type(8)))  float  v8f;

union V16U { v16bf v; v8bf h[2]; };

// hardware-rcp based activations (v_rcp_f32 + v_exp_f32, no IEEE div sequence)
__device__ __forceinline__ float sigmoid_f(float x) {
  return __builtin_amdgcn_rcpf(1.0f + __expf(-x));
}
__device__ __forceinline__ float silu_f(float x) {
  return x * sigmoid_f(x);
}

// ---------------- CDNA5 async global->LDS copy (ASYNCcnt-tracked DMA) ----------------
__device__ __forceinline__ void async_copy_b128(void* lds_dst, const void* gsrc) {
  unsigned lds_off = (unsigned)(uintptr_t)lds_dst;
  asm volatile("global_load_async_to_lds_b128 %0, %1, off"
               :: "v"(lds_off), "v"(gsrc) : "memory");
}
__device__ __forceinline__ void wait_async() {
  asm volatile("s_wait_asynccnt 0x0" ::: "memory");
}

// ---------------- WMMA fragment loaders ----------------
__device__ __forceinline__ v16bf frag_a(const __bf16* A, int ldk, int k0, int lane) {
  int row = lane & 15;
  int kb  = (lane & 16) ? 8 : 0;
  const __bf16* p = A + (size_t)row * ldk + k0 + kb;
  V16U u;
  u.h[0] = *(const v8bf*)(p);
  u.h[1] = *(const v8bf*)(p + 16);
  return u.v;
}

__device__ __forceinline__ v16bf frag_b(const __bf16* BT, int ldk, int n0, int k0, int lane) {
  int col = n0 + (lane & 15);
  int ks  = (lane & 16) ? 8 : 0;
  const __bf16* p = BT + (size_t)col * ldk + k0 + ks;
  V16U u;
  u.h[0] = *(const v8bf*)(p);
  u.h[1] = *(const v8bf*)(p + 16);
  return u.v;
}

// single-tile GEMM with next-step prefetch
template <int KP>
__device__ __forceinline__ v8f gemm_tile(const __bf16* A, const __bf16* BT,
                                         int n0, int lane) {
  v8f acc = {};
  v16bf a = frag_a(A, KP, 0, lane);
  v16bf b = frag_b(BT, KP, n0, 0, lane);
  #pragma unroll
  for (int k0 = 0; k0 < KP; k0 += 32) {
    v16bf an = a, bn = b;
    if (k0 + 32 < KP) {
      an = frag_a(A, KP, k0 + 32, lane);
      bn = frag_b(BT, KP, n0, k0 + 32, lane);
    }
    acc = __builtin_amdgcn_wmma_f32_16x16x32_bf16(false, a, false, b,
                                                  (short)0, acc, false, false);
    a = an; b = bn;
  }
  return acc;
}

// dual-col GEMM: shared A fragment feeds two N-tiles (cols n0, n0+16)
template <int KP>
__device__ __forceinline__ void gemm_tile2(const __bf16* A, const __bf16* BT,
                                           int n0, int lane, v8f& acc0, v8f& acc1) {
  v16bf a  = frag_a(A, KP, 0, lane);
  v16bf b0 = frag_b(BT, KP, n0, 0, lane);
  v16bf b1 = frag_b(BT, KP, n0 + 16, 0, lane);
  #pragma unroll
  for (int k0 = 0; k0 < KP; k0 += 32) {
    v16bf an = a, b0n = b0, b1n = b1;
    if (k0 + 32 < KP) {
      an  = frag_a(A, KP, k0 + 32, lane);
      b0n = frag_b(BT, KP, n0, k0 + 32, lane);
      b1n = frag_b(BT, KP, n0 + 16, k0 + 32, lane);
    }
    acc0 = __builtin_amdgcn_wmma_f32_16x16x32_bf16(false, a, false, b0,
                                                   (short)0, acc0, false, false);
    acc1 = __builtin_amdgcn_wmma_f32_16x16x32_bf16(false, a, false, b1,
                                                   (short)0, acc1, false, false);
    a = an; b0 = b0n; b1 = b1n;
  }
}

// 2x2 tile GEMM: two row-tiles (A rows 0-15 and 16-31) x two col-tiles share fragments.
// 4 WMMAs per (2 A + 2 B) fragment loads -> halved B traffic vs 16-row blocks.
template <int KP>
__device__ __forceinline__ void gemm_2x2(const __bf16* A /*32 x KP*/, const __bf16* BT,
                                         int n0, int lane, v8f acc[2][2]) {
  const __bf16* A1p = A + 16 * KP;
  v16bf a0 = frag_a(A,   KP, 0, lane);
  v16bf a1 = frag_a(A1p, KP, 0, lane);
  v16bf b0 = frag_b(BT, KP, n0, 0, lane);
  v16bf b1 = frag_b(BT, KP, n0 + 16, 0, lane);
  #pragma unroll
  for (int k0 = 0; k0 < KP; k0 += 32) {
    v16bf a0n = a0, a1n = a1, b0n = b0, b1n = b1;
    if (k0 + 32 < KP) {
      a0n = frag_a(A,   KP, k0 + 32, lane);
      a1n = frag_a(A1p, KP, k0 + 32, lane);
      b0n = frag_b(BT, KP, n0, k0 + 32, lane);
      b1n = frag_b(BT, KP, n0 + 16, k0 + 32, lane);
    }
    acc[0][0] = __builtin_amdgcn_wmma_f32_16x16x32_bf16(false, a0, false, b0,
                                                        (short)0, acc[0][0], false, false);
    acc[0][1] = __builtin_amdgcn_wmma_f32_16x16x32_bf16(false, a0, false, b1,
                                                        (short)0, acc[0][1], false, false);
    acc[1][0] = __builtin_amdgcn_wmma_f32_16x16x32_bf16(false, a1, false, b0,
                                                        (short)0, acc[1][0], false, false);
    acc[1][1] = __builtin_amdgcn_wmma_f32_16x16x32_bf16(false, a1, false, b1,
                                                        (short)0, acc[1][1], false, false);
    a0 = a0n; a1 = a1n; b0 = b0n; b1 = b1n;
  }
}

// ---------------- weight transpose-convert: W[K,N] f32 -> WT[N,Kp] bf16 ----------------
__global__ void k_wconv(const float* __restrict__ src, __bf16* __restrict__ dst,
                        int K, int N, int Kp) {
  int i = blockIdx.x * blockDim.x + threadIdx.x;
  if (i >= N * Kp) return;
  int n = i / Kp, k = i - n * Kp;
  dst[i] = (k < K) ? (__bf16)src[(size_t)k * N + n] : (__bf16)0.0f;
}

__global__ void k_zero2(float* __restrict__ a, float* __restrict__ b, int n) {
  int i = blockIdx.x * blockDim.x + threadIdx.x;
  if (i < n) { a[i] = 0.0f; b[i] = 0.0f; }
}

// ---------------- atom embedding: h = LN(silu(atoms @ W + b)) ----------------
__global__ void k_embed(const float* __restrict__ atoms, const __bf16* __restrict__ WT,
                        const float* __restrict__ bias, const float* __restrict__ gam,
                        const float* __restrict__ bet, float* __restrict__ h,
                        __bf16* __restrict__ hbf) {
  __shared__ __bf16 As[16 * DAP];
  __shared__ float  Cs[16 * DN];
  int m0 = blockIdx.x * 16;
  int tid = threadIdx.x;
  for (int i = tid; i < 16 * DAP; i += 256) {
    int row = i / DAP, k = i - row * DAP;
    As[i] = (k < DA) ? (__bf16)atoms[(size_t)(m0 + row) * DA + k] : (__bf16)0.0f;
  }
  __syncthreads();
  int wave = tid >> 5, lane = tid & 31;
  {
    int n0 = wave * 32;
    v8f acc0 = {}, acc1 = {};
    gemm_tile2<DAP>(As, WT, n0, lane, acc0, acc1);
    int rb = (lane & 16) ? 8 : 0;
    int col0 = n0 + (lane & 15), col1 = col0 + 16;
    float bc0 = bias[col0], bc1 = bias[col1];
    #pragma unroll
    for (int r = 0; r < 8; ++r) {
      Cs[(rb + r) * DN + col0] = silu_f(acc0[r] + bc0);
      Cs[(rb + r) * DN + col1] = silu_f(acc1[r] + bc1);
    }
  }
  __syncthreads();
  int row = tid >> 4, c0 = tid & 15;
  float s = 0.f, s2 = 0.f, x[16];
  #pragma unroll
  for (int j = 0; j < 16; ++j) {
    float v = Cs[row * DN + c0 + 16 * j];
    x[j] = v; s += v; s2 += v * v;
  }
  for (int m = 1; m < 16; m <<= 1) { s += __shfl_xor(s, m, 32); s2 += __shfl_xor(s2, m, 32); }
  float mean = s * (1.f / DN);
  float var  = s2 * (1.f / DN) - mean * mean;
  float inv  = rsqrtf(var + 1e-5f);
  #pragma unroll
  for (int j = 0; j < 16; ++j) {
    int col = c0 + 16 * j;
    float y = (x[j] - mean) * inv * gam[col] + bet[col];
    size_t idx = (size_t)(m0 + row) * DN + col;
    h[idx] = y;
    hbf[idx] = (__bf16)y;
  }
}

// ---------------- fused edge encoder ----------------
__global__ void k_edge_enc(const float* __restrict__ atoms, const float* __restrict__ edge_r,
                           const int* __restrict__ src, const int* __restrict__ dst,
                           const __bf16* __restrict__ W1T, const float* __restrict__ b1,
                           const __bf16* __restrict__ W2T, const float* __restrict__ b2,
                           const float* __restrict__ gam, const float* __restrict__ bet,
                           __bf16* __restrict__ ef) {
  __shared__ __bf16 Ein[16 * DIN_E];
  __shared__ __bf16 T1[16 * DE];
  __shared__ float  Cs[16 * DE];
  __shared__ float  distS[16];
  int e0 = blockIdx.x * 16;
  int tid = threadIdx.x;
  if (tid < 16) {
    float rx = edge_r[(size_t)(e0 + tid) * 3 + 0];
    float ry = edge_r[(size_t)(e0 + tid) * 3 + 1];
    float rz = edge_r[(size_t)(e0 + tid) * 3 + 2];
    distS[tid] = sqrtf(rx * rx + ry * ry + rz * rz);
  }
  __syncthreads();
  for (int i = tid; i < 16 * DIN_E; i += 256) {
    int row = i / DIN_E, k = i - row * DIN_E;
    int e = e0 + row;
    float v;
    if (k < 92)       v = atoms[(size_t)src[e] * DA + k];
    else if (k < 184) v = atoms[(size_t)dst[e] * DA + (k - 92)];
    else {
      int b_ = k - 184;
      float d = distS[row] - (8.0f / 39.0f) * (float)b_;  // linspace(0,8,40)
      v = __expf(-(d * d) * (1.0f / 0.04f));              // width = 8/40
    }
    Ein[i] = (__bf16)v;
  }
  __syncthreads();
  int wave = tid >> 5, lane = tid & 31;
  {
    int n0 = wave * 16;
    v8f acc = gemm_tile<DIN_E>(Ein, W1T, n0, lane);
    int col = n0 + (lane & 15), rb = (lane & 16) ? 8 : 0;
    float bc = b1[col];
    #pragma unroll
    for (int r = 0; r < 8; ++r) T1[(rb + r) * DE + col] = (__bf16)silu_f(acc[r] + bc);
  }
  __syncthreads();
  {
    int n0 = wave * 16;
    v8f acc = gemm_tile<DE>(T1, W2T, n0, lane);
    int col = n0 + (lane & 15), rb = (lane & 16) ? 8 : 0;
    float bc = b2[col];
    #pragma unroll
    for (int r = 0; r < 8; ++r) Cs[(rb + r) * DE + col] = silu_f(acc[r] + bc);
  }
  __syncthreads();
  int row = tid >> 4, c0 = tid & 15;
  float s = 0.f, s2 = 0.f, x[8];
  #pragma unroll
  for (int j = 0; j < 8; ++j) {
    float v = Cs[row * DE + c0 + 16 * j];
    x[j] = v; s += v; s2 += v * v;
  }
  for (int m = 1; m < 16; m <<= 1) { s += __shfl_xor(s, m, 32); s2 += __shfl_xor(s2, m, 32); }
  float mean = s * (1.f / DE);
  float var  = s2 * (1.f / DE) - mean * mean;
  float inv  = rsqrtf(var + 1e-5f);
  #pragma unroll
  for (int j = 0; j < 8; ++j) {
    int col = c0 + 16 * j;
    ef[(size_t)(e0 + row) * DE + col] = (__bf16)((x[j] - mean) * inv * gam[col] + bet[col]);
  }
}

// ---------------- node projection (32 rows, 2x2 tiles): out = hbf @ WT + bias ----------------
__global__ void k_nodeproj(const __bf16* __restrict__ hbf, const __bf16* __restrict__ WT,
                           const float* __restrict__ bias, float* __restrict__ out) {
  __shared__ __bf16 As[32 * DN];
  int m0 = blockIdx.x * 32, tid = threadIdx.x;
  // async DMA: 32 rows x 256 bf16 = 16KB contiguous = 1024 x b128 chunks
  const __bf16* hb = hbf + (size_t)m0 * DN;
  for (int i = tid; i < 1024; i += 256)
    async_copy_b128(&As[i * 8], hb + i * 8);
  wait_async();
  __syncthreads();
  int wave = tid >> 5, lane = tid & 31;
  int n0 = wave * 32;
  v8f acc[2][2] = {};
  gemm_2x2<DN>(As, WT, n0, lane, acc);
  int rb = (lane & 16) ? 8 : 0;
  int col0 = n0 + (lane & 15), col1 = col0 + 16;
  float bc0 = bias[col0], bc1 = bias[col1];
  #pragma unroll
  for (int rt = 0; rt < 2; ++rt) {
    #pragma unroll
    for (int r = 0; r < 8; ++r) {
      size_t rowo = (size_t)(m0 + rt * 16 + rb + r) * DN;
      out[rowo + col0] = acc[rt][0][r] + bc0;
      out[rowo + col1] = acc[rt][1][r] + bc1;
    }
  }
}

// ---------------- fused message (32 edges, 2x2 tiles) ----------------
__global__ void k_edge_msg(const __bf16* __restrict__ ef, const __bf16* __restrict__ WT,
                           const float* __restrict__ beg,
                           const int* __restrict__ src, const int* __restrict__ dst,
                           const float* __restrict__ aP, const float* __restrict__ bP,
                           const float* __restrict__ cP,
                           float* __restrict__ num, float* __restrict__ den) {
  __shared__ __bf16 As[32 * DE];
  int e0 = blockIdx.x * 32, tid = threadIdx.x;
  // async DMA: 32 rows x 128 bf16 = 8KB contiguous = 512 x b128 chunks
  const __bf16* eb = ef + (size_t)e0 * DE;
  for (int i = tid; i < 512; i += 256)
    async_copy_b128(&As[i * 8], eb + i * 8);
  wait_async();
  __syncthreads();
  int wave = tid >> 5, lane = tid & 31;
  int n0 = wave * 32;
  v8f acc[2][2] = {};
  gemm_2x2<DE>(As, WT, n0, lane, acc);
  int rb = (lane & 16) ? 8 : 0;
  int col0 = n0 + (lane & 15), col1 = col0 + 16;
  float bc0 = beg[col0], bc1 = beg[col1];
  #pragma unroll
  for (int rt = 0; rt < 2; ++rt) {
    #pragma unroll
    for (int r = 0; r < 8; ++r) {
      int e  = e0 + rt * 16 + rb + r;
      int s_ = src[e], d_ = dst[e];
      size_t so = (size_t)s_ * DN, dofs = (size_t)d_ * DN;
      float pre0 = acc[rt][0][r] + bc0 + aP[so + col0] + bP[dofs + col0];
      float pre1 = acc[rt][1][r] + bc1 + aP[so + col1] + bP[dofs + col1];
      float g0 = sigmoid_f(pre0), g1 = sigmoid_f(pre1);
      float q0 = g0 * cP[so + col0], q1 = g1 * cP[so + col1];
      atomicAdd(&num[dofs + col0], q0);
      atomicAdd(&den[dofs + col0], g0);
      atomicAdd(&num[dofs + col1], q1);
      atomicAdd(&den[dofs + col1], g1);
    }
  }
}

// ---------------- node update: h = LN(silu(self + num/(den+eps)) + h) ----------------
__global__ void k_node_update(float* __restrict__ h, __bf16* __restrict__ hbf,
                              const float* __restrict__ sP,
                              const float* __restrict__ num, const float* __restrict__ den,
                              const float* __restrict__ lng, const float* __restrict__ lnb) {
  int m0 = blockIdx.x * 16, tid = threadIdx.x;
  int row = m0 + (tid >> 4), c0 = tid & 15;
  float x[16], s = 0.f, s2 = 0.f;
  #pragma unroll
  for (int j = 0; j < 16; ++j) {
    int col = c0 + 16 * j;
    size_t idx = (size_t)row * DN + col;
    float v = sP[idx] + num[idx] * __builtin_amdgcn_rcpf(den[idx] + 1e-6f);
    v = silu_f(v);
    v += h[idx];
    x[j] = v; s += v; s2 += v * v;
  }
  for (int m = 1; m < 16; m <<= 1) { s += __shfl_xor(s, m, 32); s2 += __shfl_xor(s2, m, 32); }
  float mean = s * (1.f / DN);
  float var  = s2 * (1.f / DN) - mean * mean;
  float inv  = rsqrtf(var + 1e-5f);
  #pragma unroll
  for (int j = 0; j < 16; ++j) {
    int col = c0 + 16 * j;
    size_t idx = (size_t)row * DN + col;
    float y = (x[j] - mean) * inv * lng[col] + lnb[col];
    h[idx] = y;
    hbf[idx] = (__bf16)y;
  }
}

// ---------------- fused head MLP (32 edges): 640 -> 256 -> 128 -> 64 -> out ----------------
__global__ void k_head(const __bf16* __restrict__ ef, const __bf16* __restrict__ hbf,
                       const int* __restrict__ src, const int* __restrict__ dst,
                       const __bf16* __restrict__ W1T, const float* __restrict__ b1,
                       const __bf16* __restrict__ W2T, const float* __restrict__ b2,
                       const __bf16* __restrict__ W3T, const float* __restrict__ b3,
                       const float* __restrict__ outW, const float* __restrict__ outB,
                       float* __restrict__ out) {
  // LDS overlay: A1 (32x640 bf16 = 40KB) dead after GEMM1 -> reuse for T2/T3.
  __shared__ __align__(16) char smem[40960 + 16384];  // 56KB
  __bf16* A1 = (__bf16*)smem;                 // 32*640 bf16
  __bf16* T1 = (__bf16*)(smem + 40960);       // 32*256 bf16
  __bf16* T2 = (__bf16*)smem;                 // 32*128 bf16 (aliases A1)
  float*  T3 = (float*)(smem + 8192);         // 32*64 f32  (aliases A1 tail)
  int e0 = blockIdx.x * 32, tid = threadIdx.x;
  // gathered async DMA: row = [ef row | hbf[src] row | hbf[dst] row] = 80 b128 chunks
  for (int i = tid; i < 32 * 80; i += 256) {
    int row = i / 80, c = i - row * 80;
    int e = e0 + row;
    const __bf16* g;
    if (c < 16)      g = ef  + (size_t)e * DE + c * 8;
    else if (c < 48) g = hbf + (size_t)src[e] * DN + (c - 16) * 8;
    else             g = hbf + (size_t)dst[e] * DN + (c - 48) * 8;
    async_copy_b128(&A1[row * DH + c * 8], g);
  }
  wait_async();
  __syncthreads();
  int wave = tid >> 5, lane = tid & 31;
  {
    int n0 = wave * 32;
    v8f acc[2][2] = {};
    gemm_2x2<DH>(A1, W1T, n0, lane, acc);
    int rb = (lane & 16) ? 8 : 0;
    int col0 = n0 + (lane & 15), col1 = col0 + 16;
    float bc0 = b1[col0], bc1 = b1[col1];
    #pragma unroll
    for (int rt = 0; rt < 2; ++rt) {
      #pragma unroll
      for (int r = 0; r < 8; ++r) {
        int rowo = (rt * 16 + rb + r) * DN;
        T1[rowo + col0] = (__bf16)silu_f(acc[rt][0][r] + bc0);
        T1[rowo + col1] = (__bf16)silu_f(acc[rt][1][r] + bc1);
      }
    }
  }
  __syncthreads();
  {
    // 2 row-tiles x 8 col-tiles = 16 tiles over 8 waves: wave -> (row w>>2, cols (w&3)*32)
    int rt = wave >> 2, n0 = (wave & 3) * 32;
    v8f acc0 = {}, acc1 = {};
    gemm_tile2<DN>(T1 + rt * 16 * DN, W2T, n0, lane, acc0, acc1);
    int rb = (lane & 16) ? 8 : 0;
    int col0 = n0 + (lane & 15), col1 = col0 + 16;
    float bc0 = b2[col0], bc1 = b2[col1];
    #pragma unroll
    for (int r = 0; r < 8; ++r) {
      int rowo = (rt * 16 + rb + r) * DE;
      T2[rowo + col0] = (__bf16)silu_f(acc0[r] + bc0);
      T2[rowo + col1] = (__bf16)silu_f(acc1[r] + bc1);
    }
  }
  __syncthreads();
  {
    // 2 row-tiles x 4 col-tiles = 8 tiles over 8 waves
    int rt = wave >> 2, n0 = (wave & 3) * 16;
    v8f acc = gemm_tile<DE>(T2 + rt * 16 * DE, W3T, n0, lane);
    int col = n0 + (lane & 15), rb = (lane & 16) ? 8 : 0;
    float bc = b3[col];
    #pragma unroll
    for (int r = 0; r < 8; ++r) T3[(rt * 16 + rb + r) * 64 + col] = silu_f(acc[r] + bc);
  }
  __syncthreads();
  if (tid < 32) {
    float acc = outB[0];
    #pragma unroll
    for (int k = 0; k < 64; ++k) acc += T3[tid * 64 + k] * outW[k * 2 + 0];
    out[e0 + tid] = acc;
  }
}

// ---------------- host launcher ----------------
extern "C" void kernel_launch(void* const* d_in, const int* in_sizes, int n_in,
                              void* d_out, int out_size, void* d_ws, size_t ws_size,
                              hipStream_t stream) {
  (void)in_sizes; (void)n_in; (void)out_size; (void)ws_size;
  const float* atoms  = (const float*)d_in[0];
  const float* edge_r = (const float*)d_in[1];
  const int*   src    = (const int*)d_in[2];
  const int*   dst    = (const int*)d_in[3];
  const float* emb_W  = (const float*)d_in[4];
  const float* emb_b  = (const float*)d_in[5];
  const float* emb_g  = (const float*)d_in[6];
  const float* emb_be = (const float*)d_in[7];
  const float* ee_W1  = (const float*)d_in[8];
  const float* ee_b1  = (const float*)d_in[9];
  const float* ee_W2  = (const float*)d_in[10];
  const float* ee_b2  = (const float*)d_in[11];
  const float* ee_g   = (const float*)d_in[12];
  const float* ee_be  = (const float*)d_in[13];
  const float* g_Wsg  = (const float*)d_in[14];
  const float* g_bsg  = (const float*)d_in[15];
  const float* g_Wdg  = (const float*)d_in[16];
  const float* g_bdg  = (const float*)d_in[17];
  const float* g_Weg  = (const float*)d_in[18];
  const float* g_beg  = (const float*)d_in[19];
  const float* g_Wmsg = (const float*)d_in[20];
  const float* g_bmsg = (const float*)d_in[21];
  const float* g_Wself= (const float*)d_in[22];
  const float* g_bself= (const float*)d_in[23];
  const float* ln_g   = (const float*)d_in[24];
  const float* ln_b   = (const float*)d_in[25];
  const float* p_W1   = (const float*)d_in[26];
  const float* p_b1   = (const float*)d_in[27];
  const float* p_W2   = (const float*)d_in[28];
  const float* p_b2   = (const float*)d_in[29];
  const float* p_W3   = (const float*)d_in[30];
  const float* p_b3   = (const float*)d_in[31];
  const float* out_W  = (const float*)d_in[32];
  const float* out_b  = (const float*)d_in[33];
  float* out = (float*)d_out;

  // workspace carving
  char* ws = (char*)d_ws;
  size_t off = 0;
  auto carve = [&](size_t bytes) -> char* {
    char* p = ws + off;
    off += (bytes + 255) & ~(size_t)255;
    return p;
  };
  const size_t nodeF = (size_t)NNODES * DN * sizeof(float);
  float* h    = (float*)carve(nodeF);
  float* aP   = (float*)carve(nodeF);
  float* bP   = (float*)carve(nodeF);
  float* cP   = (float*)carve(nodeF);
  float* sP   = (float*)carve(nodeF);
  float* numB = (float*)carve(nodeF);
  float* denB = (float*)carve(nodeF);
  __bf16* hbf  = (__bf16*)carve((size_t)NNODES * DN * 2);
  __bf16* ef   = (__bf16*)carve((size_t)NEDGES * DE * 2);
  __bf16* embT = (__bf16*)carve((size_t)DN * DAP * 2);
  __bf16* ee1T = (__bf16*)carve((size_t)DE * DIN_E * 2);
  __bf16* ee2T = (__bf16*)carve((size_t)DE * DE * 2);
  __bf16* sgT  = (__bf16*)carve((size_t)NLAYERS * DN * DN * 2);
  __bf16* dgT  = (__bf16*)carve((size_t)NLAYERS * DN * DN * 2);
  __bf16* msT  = (__bf16*)carve((size_t)NLAYERS * DN * DN * 2);
  __bf16* slT  = (__bf16*)carve((size_t)NLAYERS * DN * DN * 2);
  __bf16* egT  = (__bf16*)carve((size_t)NLAYERS * DN * DE * 2);
  __bf16* p1T  = (__bf16*)carve((size_t)DN * DH * 2);
  __bf16* p2T  = (__bf16*)carve((size_t)DE * DN * 2);
  __bf16* p3T  = (__bf16*)carve((size_t)64 * DE * 2);

  auto wconv = [&](const float* s, __bf16* d, int K, int N, int Kp) {
    int total = N * Kp;
    k_wconv<<<(total + 255) / 256, 256, 0, stream>>>(s, d, K, N, Kp);
  };

  // weight transpose-convert (fp32 row-major [K,N] -> bf16 col-major [N,Kp])
  wconv(emb_W, embT, DA, DN, DAP);
  wconv(ee_W1, ee1T, DIN_E, DE, DIN_E);
  wconv(ee_W2, ee2T, DE, DE, DE);
  for (int l = 0; l < NLAYERS; ++l) {
    wconv(g_Wsg  + (size_t)l * DN * DN, sgT + (size_t)l * DN * DN, DN, DN, DN);
    wconv(g_Wdg  + (size_t)l * DN * DN, dgT + (size_t)l * DN * DN, DN, DN, DN);
    wconv(g_Wmsg + (size_t)l * DN * DN, msT + (size_t)l * DN * DN, DN, DN, DN);
    wconv(g_Wself+ (size_t)l * DN * DN, slT + (size_t)l * DN * DN, DN, DN, DN);
    wconv(g_Weg  + (size_t)l * DE * DN, egT + (size_t)l * DN * DE, DE, DN, DE);
  }
  wconv(p_W1, p1T, DH, DN, DH);
  wconv(p_W2, p2T, DN, DE, DN);
  wconv(p_W3, p3T, DE, 64, DE);

  const int nodeBlocks16 = NNODES / 16;   // 1250
  const int nodeBlocks32 = NNODES / 32;   // 625
  const int edgeBlocks16 = NEDGES / 16;   // 20000
  const int edgeBlocks32 = NEDGES / 32;   // 10000

  // atom embedding + edge encoder (independent)
  k_embed<<<nodeBlocks16, 256, 0, stream>>>(atoms, embT, emb_b, emb_g, emb_be, h, hbf);
  k_edge_enc<<<edgeBlocks16, 256, 0, stream>>>(atoms, edge_r, src, dst,
                                               ee1T, ee_b1, ee2T, ee_b2, ee_g, ee_be, ef);

  // edge-gated graph conv layers
  for (int l = 0; l < NLAYERS; ++l) {
    k_nodeproj<<<nodeBlocks32, 256, 0, stream>>>(hbf, sgT + (size_t)l * DN * DN, g_bsg  + l * DN, aP);
    k_nodeproj<<<nodeBlocks32, 256, 0, stream>>>(hbf, dgT + (size_t)l * DN * DN, g_bdg  + l * DN, bP);
    k_nodeproj<<<nodeBlocks32, 256, 0, stream>>>(hbf, msT + (size_t)l * DN * DN, g_bmsg + l * DN, cP);
    k_nodeproj<<<nodeBlocks32, 256, 0, stream>>>(hbf, slT + (size_t)l * DN * DN, g_bself+ l * DN, sP);
    int nElems = NNODES * DN;
    k_zero2<<<(nElems + 255) / 256, 256, 0, stream>>>(numB, denB, nElems);
    k_edge_msg<<<edgeBlocks32, 256, 0, stream>>>(ef, egT + (size_t)l * DN * DE, g_beg + l * DN,
                                                 src, dst, aP, bP, cP, numB, denB);
    k_node_update<<<nodeBlocks16, 256, 0, stream>>>(h, hbf, sP, numB, denB,
                                                    ln_g + l * DN, ln_b + l * DN);
  }

  // head MLP
  k_head<<<edgeBlocks32, 256, 0, stream>>>(ef, hbf, src, dst,
                                           p1T, p_b1, p2T, p_b2, p3T, p_b3,
                                           out_W, out_b, out);
}
